// ScaledDotProductAttention_22282290331874
// MI455X (gfx1250) — matrix operand order repaired and checked by reading, not compiled, not run on hardware
//
#include <hip/hip_runtime.h>

typedef __attribute__((ext_vector_type(16))) _Float16 v16h;
typedef __attribute__((ext_vector_type(8)))  _Float16 v8h;
typedef __attribute__((ext_vector_type(8)))  float    v8f;

#define N_TOK  4096
#define DIM    1024
#define BR     16          // query rows per block
#define BC     32          // keys per tile (WMMA K for the PV matmul)
#define NWAVE  8
#define WSLICE 128         // DIM / NWAVE, D-slice owned by each wave
#define NTILE  (N_TOK / BC)

// ---------------------------------------------------------------------------
// Prep kernel 1: K (f32, row-major) -> f16 row-major in workspace.
// ---------------------------------------------------------------------------
__launch_bounds__(256)
__global__ void cvt_k_kernel(const float* __restrict__ kg, _Float16* __restrict__ kh)
{
    const size_t i = (size_t)blockIdx.x * blockDim.x + threadIdx.x;  // over N*D/8
    const float4 a = ((const float4*)kg)[2 * i + 0];
    const float4 b = ((const float4*)kg)[2 * i + 1];
    v8h o;
    o[0] = (_Float16)a.x; o[1] = (_Float16)a.y; o[2] = (_Float16)a.z; o[3] = (_Float16)a.w;
    o[4] = (_Float16)b.x; o[5] = (_Float16)b.y; o[6] = (_Float16)b.z; o[7] = (_Float16)b.w;
    ((v8h*)kh)[i] = o;
}

// ---------------------------------------------------------------------------
// Prep kernel 2: V (f32, [n][d]) -> f16 transposed [d][n] via LDS tile.
// ---------------------------------------------------------------------------
#define TP 64
__launch_bounds__(256)
__global__ void cvt_vt_kernel(const float* __restrict__ vg, _Float16* __restrict__ vth)
{
    __shared__ _Float16 tile[TP][TP + 8];          // pad: avoid bank conflicts
    const int n0 = blockIdx.x * TP;                // token block
    const int d0 = blockIdx.y * TP;                // dim block
    const int tid = threadIdx.x;

    for (int i = tid; i < TP * TP / 4; i += 256) { // coalesced f32 reads over d
        const int r  = i >> 4;                     // n within tile
        const int c4 = i & 15;                     // d/4 within tile
        const float4 f = ((const float4*)vg)[((size_t)(n0 + r) * DIM + d0) / 4 + c4];
        const int c = c4 * 4;
        tile[c + 0][r] = (_Float16)f.x;
        tile[c + 1][r] = (_Float16)f.y;
        tile[c + 2][r] = (_Float16)f.z;
        tile[c + 3][r] = (_Float16)f.w;
    }
    __syncthreads();
    for (int i = tid; i < TP * TP / 8; i += 256) { // coalesced f16 writes over n
        const int dr = i >> 3;
        const int c8 = (i & 7) * 8;
        v8h o;
        #pragma unroll
        for (int jj = 0; jj < 8; ++jj) o[jj] = tile[dr][c8 + jj];
        *(v8h*)&vth[(size_t)(d0 + dr) * N_TOK + n0 + c8] = o;
    }
}

// ---------------------------------------------------------------------------
// Main flash-attention kernel. K/V come pre-converted to f16 (Vt transposed),
// so both WMMA B-operands are single 32-byte contiguous global loads per lane.
// Two barriers per key tile; softmax split across 32 lanes of wave 0; O-rescale
// skipped via a scalar branch whenever the running row max did not change.
// ---------------------------------------------------------------------------
__launch_bounds__(256)
__global__ void fa_kernel(const float* __restrict__ qg,
                          const _Float16* __restrict__ kh,   // [n][d] f16
                          const _Float16* __restrict__ vth,  // [d][n] f16
                          float* __restrict__ og)
{
    __shared__ __align__(64) float    s_part[BR][NWAVE][BC]; // 16 KB per-wave S partials
    __shared__ __align__(64) _Float16 p_a[32][16];           // 1 KB  P in WMMA A-layout
    __shared__ float row_max[BR], row_sum[BR], row_alpha[BR];
    __shared__ int   s_flag;                                  // block-wide rescale flag

    const int tid  = threadIdx.x;
    const int lane = tid & 31;
    const int wave = tid >> 5;
    const int hi   = lane >> 4;       // half-wave (K-half select in this code's contiguous k order)
    const int mrow = lane & 15;
    const int Mb   = hi * 8;          // C-layout row base for this half-wave
    const int row0 = blockIdx.x * BR;

    // ---- Q block -> registers, pre-swizzled into the 16-bit A-operand layout ----
    // lanes 0-15 hold K = {0..15}; lanes 16-31 hold K = {16..31}
    v16h aq[4];
    {
        const float* qrow = qg + (size_t)(row0 + mrow) * DIM;
        #pragma unroll
        for (int dsl = 0; dsl < 4; ++dsl) {
            const int d0 = (wave * 4 + dsl) * 32;
            v16h a;
            #pragma unroll
            for (int h = 0; h < 16; ++h) {
                const int kk = h + (hi << 4);
                a[h] = (_Float16)qrow[d0 + kk];
            }
            aq[dsl] = a;
        }
    }

    if (tid < BR) { row_max[tid] = -3.0e38f; row_sum[tid] = 0.0f; }

    v8f o_acc[8];
    #pragma unroll
    for (int t = 0; t < 8; ++t) o_acc[t] = {};

    for (int j = 0; j < NTILE; ++j) {
        const int key0 = j * BC;

        // ---- S = Q·K^T partial over this wave's 128-wide D slice ----
        v8f s0 = {}, s1 = {};
        const _Float16* kb = kh + (size_t)(key0 + mrow) * DIM + wave * WSLICE + (hi << 4);
        #pragma unroll
        for (int dsl = 0; dsl < 4; ++dsl) {
            const v16h b0 = *(const v16h*)(kb + dsl * 32);                    // keys 0..15
            const v16h b1 = *(const v16h*)(kb + (size_t)16 * DIM + dsl * 32); // keys 16..31
            s0 = __builtin_amdgcn_wmma_f32_16x16x32_f16(false, aq[dsl], false, b0,
                                                        (short)0, s0, false, false);
            s1 = __builtin_amdgcn_wmma_f32_16x16x32_f16(false, aq[dsl], false, b1,
                                                        (short)0, s1, false, false);
        }
        #pragma unroll
        for (int r = 0; r < 8; ++r) {
            s_part[Mb + r][wave][mrow]      = s0[r];
            s_part[Mb + r][wave][mrow + 16] = s1[r];
        }

        // ---- preload PV B-operands (independent of softmax; hides latency) ----
        v16h bv[8];
        #pragma unroll
        for (int t = 0; t < 8; ++t) {
            const int dcol = wave * WSLICE + t * 16 + mrow;
            bv[t] = *(const v16h*)(vth + (size_t)dcol * N_TOK + key0 + (hi << 4));
        }

        if (j + 1 < NTILE) {  // warm caches for next K tile: global_prefetch_b8
            __builtin_prefetch(kh + (size_t)(key0 + BC) * DIM + tid * 128, 0, 1);
            __builtin_prefetch(vth + (size_t)(tid * 4) * N_TOK + key0 + BC, 0, 1);
        }
        __syncthreads();   // s_part complete; prev-iter PV done with p_a/row_alpha

        // ---- online softmax: 2 lanes per row (16 cols each), shfl combine ----
        if (wave == 0) {
            const int m  = mrow;
            const int c0 = hi << 4;              // column half handled by this lane
            float sf[16];
            #pragma unroll
            for (int cg = 0; cg < 4; ++cg) {
                float ax = 0.f, ay = 0.f, az = 0.f, aw = 0.f;
                #pragma unroll
                for (int w = 0; w < NWAVE; ++w) {
                    const float4 f = *(const float4*)&s_part[m][w][c0 + cg * 4];
                    ax += f.x; ay += f.y; az += f.z; aw += f.w;
                }
                sf[cg * 4 + 0] = ax; sf[cg * 4 + 1] = ay;
                sf[cg * 4 + 2] = az; sf[cg * 4 + 3] = aw;
            }
            float lm = sf[0];
            #pragma unroll
            for (int c = 1; c < 16; ++c) lm = fmaxf(lm, sf[c]);
            const float mx   = row_max[m];
            const float newm = fmaxf(fmaxf(lm, __shfl_xor(lm, 16)), mx);
            float lsum = 0.f;
            #pragma unroll
            for (int c = 0; c < 16; ++c) {
                const float p = __expf(sf[c] - newm);
                lsum += p;
                const int cc = c0 + c;           // scatter into WMMA A-layout
                const int al = m + ((cc >> 4) << 4);
                const int ah = cc & 15;
                p_a[al][ah] = (_Float16)p;
            }
            const float tot = lsum + __shfl_xor(lsum, 16);
            const unsigned long long bal = __ballot(newm > mx);
            if (lane == 0) s_flag = (bal != 0ull) ? 1 : 0;
            if (hi == 0) {
                const float alpha = __expf(mx - newm);
                row_sum[m]   = row_sum[m] * alpha + tot;
                row_max[m]   = newm;
                row_alpha[m] = alpha;
            }
        }
        __syncthreads();   // p_a / row_alpha / s_flag ready

        // ---- rescale O only when the running max moved (scalar branch) ----
        const int doscale = __builtin_amdgcn_readfirstlane(s_flag);
        const v16h ap = *(const v16h*)&p_a[lane][0];
        if (doscale) {
            float alr[8];
            #pragma unroll
            for (int r = 0; r < 8; ++r) alr[r] = row_alpha[Mb + r];
            #pragma unroll
            for (int t = 0; t < 8; ++t) {
                #pragma unroll
                for (int r = 0; r < 8; ++r) o_acc[t][r] *= alr[r];
            }
        }
        // ---- O += P·V over this wave's D slice (8 independent WMMA chains) ----
        #pragma unroll
        for (int t = 0; t < 8; ++t)
            o_acc[t] = __builtin_amdgcn_wmma_f32_16x16x32_f16(false, ap, false, bv[t],
                                                              (short)0, o_acc[t], false, false);
    }

    // ---- epilogue: normalize by softmax denominator and store ----
    {
        float invl[8];
        #pragma unroll
        for (int r = 0; r < 8; ++r) invl[r] = 1.0f / row_sum[Mb + r];
        #pragma unroll
        for (int t = 0; t < 8; ++t) {
            const int col = wave * WSLICE + t * 16 + mrow;
            #pragma unroll
            for (int r = 0; r < 8; ++r)
                og[(size_t)(row0 + Mb + r) * DIM + col] = o_acc[t][r] * invl[r];
        }
    }
}

extern "C" void kernel_launch(void* const* d_in, const int* in_sizes, int n_in,
                              void* d_out, int out_size, void* d_ws, size_t ws_size,
                              hipStream_t stream) {
    (void)in_sizes; (void)n_in; (void)out_size; (void)ws_size;
    const float* q = (const float*)d_in[0];
    const float* k = (const float*)d_in[1];
    const float* v = (const float*)d_in[2];
    float* out = (float*)d_out;

    _Float16* kh  = (_Float16*)d_ws;                       // 8 MB  f16 K  [n][d]
    _Float16* vth = kh + (size_t)N_TOK * DIM;              // 8 MB  f16 V^T [d][n]

    cvt_k_kernel<<<dim3((N_TOK * DIM / 8) / 256), dim3(256), 0, stream>>>(k, kh);
    cvt_vt_kernel<<<dim3(N_TOK / TP, DIM / TP), dim3(256), 0, stream>>>(v, vth);
    fa_kernel<<<dim3(N_TOK / BR), dim3(256), 0, stream>>>(q, kh, vth, out);
}
